// MultiHeadDifferentialAttention_14688788152714
// MI455X (gfx1250) — compile-verified, matrix operand order checked by reading, not run on hardware
//
#include <hip/hip_runtime.h>
#include <hip/hip_bf16.h>

#define SEQ_N     2048
#define DMODEL    2048
#define NHEADS    16
#define DHEAD     128
#define DH2       256
#define BATCH     2
#define LAMB_INIT 0.8f
#define RMS_EPS   1e-5f
#define SCALING   0.08838834764831845f  /* 1/sqrt(128) */

typedef __attribute__((ext_vector_type(16))) _Float16 v16h;
typedef __attribute__((ext_vector_type(8)))  _Float16 v8h;
typedef __attribute__((ext_vector_type(4)))  _Float16 v4h;
typedef __attribute__((ext_vector_type(8)))  float    v8f;

// ---------------------------------------------------------------------------
// Fragment loader for v_wmma_f32_16x16x32_f16 operands from a row-major
// matrix (global or LDS). Per CDNA5 ISA 16-bit A/B layout: lane r=lane&15 is
// the row (or B column), half h=lane>>4 selects K chunks:
//   elems 0..7  -> K = 8h..8h+7,  elems 8..15 -> K = 16+8h..23+8h.
// ---------------------------------------------------------------------------
static __device__ __forceinline__ v16h load_frag_rm(const _Float16* base, int ld) {
  int lane = threadIdx.x & 31;
  int r = lane & 15, h = lane >> 4;
  const v8h a = *(const v8h*)(base + (size_t)r * ld + 8 * h);
  const v8h b = *(const v8h*)(base + (size_t)r * ld + 16 + 8 * h);
  v16h f;
#pragma unroll
  for (int i = 0; i < 8; ++i) { f[i] = a[i]; f[i + 8] = b[i]; }
  return f;
}

// Async DMA of one 16-byte chunk global -> LDS (CDNA5, ASYNCcnt-tracked).
// LDS aperture keeps the byte offset in flat addr[31:0] (ISA 10.2).
static __device__ __forceinline__ void async_g2l_b128(const _Float16* g, _Float16* l) {
  unsigned lds_addr = (unsigned)(unsigned long long)l;
  asm volatile("global_load_async_to_lds_b128 %0, %1, off"
               :: "v"(lds_addr), "v"(g) : "memory");
}

// ---------------------------------------------------------------------------
// f32 -> f16 conversion, 4 elements/thread (b128 in, b64 out)
// ---------------------------------------------------------------------------
__global__ void f32_to_f16(const float* __restrict__ s, _Float16* __restrict__ d, int n4) {
  int i = blockIdx.x * blockDim.x + threadIdx.x;
  int stride = gridDim.x * blockDim.x;
  for (; i < n4; i += stride) {
    float4 v = ((const float4*)s)[i];
    v4h o;
    o[0] = (_Float16)v.x; o[1] = (_Float16)v.y;
    o[2] = (_Float16)v.z; o[3] = (_Float16)v.w;
    ((v4h*)d)[i] = o;
  }
}

// ---------------------------------------------------------------------------
// lambda = exp(lq1.lk1) - exp(lq2.lk2) + 0.8   (one wave)
// ---------------------------------------------------------------------------
__global__ void lam_kernel(const float* __restrict__ lq1, const float* __restrict__ lk1,
                           const float* __restrict__ lq2, const float* __restrict__ lk2,
                           float* __restrict__ lam_out) {
  int lane = threadIdx.x & 31;
  float d1 = 0.f, d2 = 0.f;
#pragma unroll
  for (int j = 0; j < 4; ++j) {
    int i = lane + 32 * j;
    d1 += lq1[i] * lk1[i];
    d2 += lq2[i] * lk2[i];
  }
#pragma unroll
  for (int m = 1; m < 32; m <<= 1) {
    d1 += __shfl_xor(d1, m, 32);
    d2 += __shfl_xor(d2, m, 32);
  }
  if (lane == 0) *lam_out = __expf(d1) - __expf(d2) + LAMB_INIT;
}

// ---------------------------------------------------------------------------
// WMMA GEMM, async-LDS double buffered, branch-free steady state.
//   C(M,Nout) = A(M,K) * Bm(Nout,K)^T
// mode 0: f16 scatter (B,H,SEQ_N,256)  mode 2: f16 scatter (B,H,256,SEQ_N)
// mode 1: f32 row-major (ldc)
// ---------------------------------------------------------------------------
__global__ __launch_bounds__(256)
void gemm_bt_wmma(const _Float16* __restrict__ A, const _Float16* __restrict__ Bm,
                  void* __restrict__ Cout, int K, int mode, int ldc) {
  __shared__ __align__(16) _Float16 lA[2][128 * 32];
  __shared__ __align__(16) _Float16 lB[2][128 * 32];

  const int tileN = blockIdx.x * 128;
  const int tileM = blockIdx.y * 128;
  const int wave = threadIdx.x >> 5;
  const int lane = threadIdx.x & 31;
  const int r = lane & 15, h = lane >> 4;
  const int row0 = tileM + wave * 16;

  auto issue_tile = [&](int buf, int k) {
#pragma unroll
    for (int c = 0; c < 2; ++c) {
      int ch = threadIdx.x + 256 * c;
      int row = ch >> 2, co = (ch & 3) * 8;
      async_g2l_b128(A + (size_t)(tileM + row) * K + k + co, &lA[buf][ch * 8]);
      async_g2l_b128(Bm + (size_t)(tileN + row) * K + k + co, &lB[buf][ch * 8]);
    }
  };

  v8f acc[8];
#pragma unroll
  for (int j = 0; j < 8; ++j) acc[j] = (v8f)(0.f);

  auto compute_tile = [&](int buf) {
    v16h af = load_frag_rm(&lA[buf][(wave * 16) * 32], 32);
    v16h bf[8];
#pragma unroll
    for (int j = 0; j < 8; ++j) bf[j] = load_frag_rm(&lB[buf][(16 * j) * 32], 32);
#pragma unroll
    for (int j = 0; j < 8; ++j)
      acc[j] = __builtin_amdgcn_wmma_f32_16x16x32_f16(
          false, af, false, bf[j], (short)0, acc[j], false, false);
  };

  const int nkt = K / 32;
  issue_tile(0, 0);
  for (int kt = 0; kt < nkt - 1; ++kt) {
    issue_tile((kt + 1) & 1, (kt + 1) * 32);
    asm volatile("s_wait_asynccnt 0x4" ::: "memory");  // tile kt landed
    __syncthreads();
    compute_tile(kt & 1);
    __syncthreads();  // this buffer becomes DMA target next iteration
  }
  asm volatile("s_wait_asynccnt 0x0" ::: "memory");
  __syncthreads();
  compute_tile((nkt - 1) & 1);

  if (mode == 1) {
    float* C = (float*)Cout;
#pragma unroll
    for (int j = 0; j < 8; ++j) {
      int col = tileN + 16 * j + r;
#pragma unroll
      for (int v = 0; v < 8; ++v) {
        int row = row0 + v + 8 * h;
        C[(size_t)row * ldc + col] = acc[j][v];
      }
    }
  } else {
    _Float16* C = (_Float16*)Cout;
#pragma unroll
    for (int j = 0; j < 8; ++j) {
      int col = tileN + 16 * j + r;       // col: head = col>>8, d = col&255
      int hh = col >> 8, d = col & 255;
#pragma unroll
      for (int v = 0; v < 8; ++v) {
        int row = row0 + v + 8 * h;       // row: b = row>>11, n = row&2047
        int b = row >> 11, n = row & (SEQ_N - 1);
        size_t bh = (size_t)(b * NHEADS + hh);
        if (mode == 0) {
          C[(bh * SEQ_N + n) * DH2 + d] = (_Float16)acc[j][v];
        } else {  // mode 2: V transposed (B,H,256,SEQ_N)
          C[(bh * DH2 + d) * SEQ_N + n] = (_Float16)acc[j][v];
        }
      }
    }
  }
}

// ---------------------------------------------------------------------------
// Flash attention, one softmax component (blockIdx.z: 0 -> Q1K1, 1 -> Q2K2).
// Grid: (B*H, SEQ_N/128, 2). Block 256 = 8 waves; wave owns 16 query rows.
// K (64x128) and VT (256x64) tiles are async-DMA'd into double-buffered LDS
// once per block and shared by all waves. LDS: 2*48KB + 16KB = 112KB.
// ---------------------------------------------------------------------------
__global__ __launch_bounds__(256)
void flash_component(const _Float16* __restrict__ Qall, const _Float16* __restrict__ Kall,
                     const _Float16* __restrict__ VT,
                     float* __restrict__ O1, float* __restrict__ O2) {
  __shared__ __align__(16) _Float16 lK[2][64 * 128];
  __shared__ __align__(16) _Float16 lV[2][256 * 64];
  __shared__ __align__(16) _Float16 ldsP[8 * 16 * 64];  // per-wave 16x64 P tile

  const int bh   = blockIdx.x;
  const int comp = blockIdx.z;
  const int doff = comp * DHEAD;
  float* Oout = comp ? O2 : O1;

  const int wave = threadIdx.x >> 5;
  const int lane = threadIdx.x & 31;
  const int r = lane & 15, h = lane >> 4;
  const int row0 = blockIdx.y * 128 + wave * 16;

  const _Float16* Qb = Qall + (size_t)bh * SEQ_N * DH2;
  const _Float16* Kb = Kall + (size_t)bh * SEQ_N * DH2;
  const _Float16* Vb = VT + (size_t)bh * DH2 * SEQ_N;

  v16h qf[4];
#pragma unroll
  for (int kk = 0; kk < 4; ++kk)
    qf[kk] = load_frag_rm(Qb + (size_t)row0 * DH2 + doff + 32 * kk, DH2);

  v8f o[16];
#pragma unroll
  for (int ct = 0; ct < 16; ++ct) o[ct] = (v8f)(0.f);
  float m_i[8], l_i[8];
#pragma unroll
  for (int v = 0; v < 8; ++v) { m_i[v] = -1e30f; l_i[v] = 0.f; }

  _Float16* wp = ldsP + wave * (16 * 64);

  // DMA one 64-key tile: K 64x128 (1024 chunks) + VT 256x64 (2048 chunks),
  // 12 async b128 per thread.
  auto issue_tile = [&](int buf, int kbase) {
#pragma unroll
    for (int c = 0; c < 4; ++c) {
      int ch = threadIdx.x + 256 * c;
      int row = ch >> 4, co = (ch & 15) * 8;
      async_g2l_b128(Kb + (size_t)(kbase + row) * DH2 + doff + co, &lK[buf][ch * 8]);
    }
#pragma unroll
    for (int c = 0; c < 8; ++c) {
      int ch = threadIdx.x + 256 * c;
      int row = ch >> 3, co = (ch & 7) * 8;
      async_g2l_b128(Vb + (size_t)row * SEQ_N + kbase + co, &lV[buf][ch * 8]);
    }
  };

  auto compute_tile = [&](int buf) {
    // S = Q * K^T (16 queries x 64 keys), K fragments from LDS
    v8f s[4];
#pragma unroll
    for (int s_ = 0; s_ < 4; ++s_) s[s_] = (v8f)(0.f);
#pragma unroll
    for (int s_ = 0; s_ < 4; ++s_) {
      v16h kf[4];
#pragma unroll
      for (int kk = 0; kk < 4; ++kk)
        kf[kk] = load_frag_rm(&lK[buf][(16 * s_) * 128 + 32 * kk], 128);
#pragma unroll
      for (int kk = 0; kk < 4; ++kk)
        s[s_] = __builtin_amdgcn_wmma_f32_16x16x32_f16(
            false, qf[kk], false, kf[kk], (short)0, s[s_], false, false);
    }
#pragma unroll
    for (int s_ = 0; s_ < 4; ++s_)
#pragma unroll
      for (int v = 0; v < 8; ++v) s[s_][v] *= SCALING;

    // online softmax: row = v + 8*h lives within the lane's 16-lane half
    float mn[8], al[8];
#pragma unroll
    for (int v = 0; v < 8; ++v) {
      float x = fmaxf(fmaxf(s[0][v], s[1][v]), fmaxf(s[2][v], s[3][v]));
#pragma unroll
      for (int msk = 1; msk < 16; msk <<= 1) x = fmaxf(x, __shfl_xor(x, msk, 32));
      mn[v] = fmaxf(m_i[v], x);
      al[v] = __expf(m_i[v] - mn[v]);
      m_i[v] = mn[v];
    }
    float lt[8];
#pragma unroll
    for (int v = 0; v < 8; ++v) lt[v] = 0.f;
#pragma unroll
    for (int s_ = 0; s_ < 4; ++s_) {
#pragma unroll
      for (int v = 0; v < 8; ++v) {
        float p = __expf(s[s_][v] - mn[v]);
        s[s_][v] = p;
        lt[v] += p;
      }
    }
#pragma unroll
    for (int v = 0; v < 8; ++v) {
      float x = lt[v];
#pragma unroll
      for (int msk = 1; msk < 16; msk <<= 1) x += __shfl_xor(x, msk, 32);
      l_i[v] = l_i[v] * al[v] + x;
    }
#pragma unroll
    for (int ct = 0; ct < 16; ++ct)
#pragma unroll
      for (int v = 0; v < 8; ++v) o[ct][v] *= al[v];

    // C-layout -> A-layout re-fragmentation via per-wave LDS region
#pragma unroll
    for (int s_ = 0; s_ < 4; ++s_)
#pragma unroll
      for (int v = 0; v < 8; ++v)
        wp[(v + 8 * h) * 64 + s_ * 16 + r] = (_Float16)s[s_][v];

    v16h pf[2];
#pragma unroll
    for (int kk = 0; kk < 2; ++kk) pf[kk] = load_frag_rm(wp + 32 * kk, 64);

    // O += P * V, V fragments from LDS (VT layout keeps them contiguous)
#pragma unroll
    for (int ct = 0; ct < 16; ++ct) {
      v16h vf[2];
#pragma unroll
      for (int kk = 0; kk < 2; ++kk)
        vf[kk] = load_frag_rm(&lV[buf][(16 * ct) * 64 + 32 * kk], 64);
#pragma unroll
      for (int kk = 0; kk < 2; ++kk)
        o[ct] = __builtin_amdgcn_wmma_f32_16x16x32_f16(
            false, pf[kk], false, vf[kk], (short)0, o[ct], false, false);
    }
  };

  const int NT = SEQ_N / 64;
  issue_tile(0, 0);
  for (int t = 0; t < NT - 1; ++t) {
    issue_tile((t + 1) & 1, (t + 1) * 64);
    asm volatile("s_wait_asynccnt 0xc" ::: "memory");  // tile t landed (12 newer pending)
    __syncthreads();
    compute_tile(t & 1);
    __syncthreads();
  }
  asm volatile("s_wait_asynccnt 0x0" ::: "memory");
  __syncthreads();
  compute_tile((NT - 1) & 1);

  float inv[8];
#pragma unroll
  for (int v = 0; v < 8; ++v) inv[v] = 1.f / l_i[v];
#pragma unroll
  for (int ct = 0; ct < 16; ++ct) {
    int d = 16 * ct + r;
#pragma unroll
    for (int v = 0; v < 8; ++v) {
      int row = row0 + v + 8 * h;
      Oout[((size_t)bh * SEQ_N + row) * DH2 + d] = o[ct][v] * inv[v];
    }
  }
}

// ---------------------------------------------------------------------------
// O = O1 - lam*O2 ; RMS-norm over 256 dims ; * rms_scale * (1-0.8) ; -> f16
// ---------------------------------------------------------------------------
__global__ __launch_bounds__(256)
void combine_rms(const float* __restrict__ O1, const float* __restrict__ O2,
                 const float* __restrict__ lam_p, const float* __restrict__ rms_scale,
                 _Float16* __restrict__ Ocat) {
  const int g = blockIdx.x * 8 + (threadIdx.x >> 5);
  const int lane = threadIdx.x & 31;
  const int hh = g & 15;
  const int bn = g >> 4;              // b*2048 + n
  const float lam = *lam_p;

  const size_t base = ((size_t)(((bn >> 11) * NHEADS) + hh) * SEQ_N + (bn & (SEQ_N - 1))) * DH2;
  float vals[8], ss = 0.f;
#pragma unroll
  for (int i = 0; i < 8; ++i) {
    int d = lane * 8 + i;
    float v = O1[base + d] - lam * O2[base + d];
    vals[i] = v;
    ss += v * v;
  }
#pragma unroll
  for (int msk = 1; msk < 32; msk <<= 1) ss += __shfl_xor(ss, msk, 32);
  const float inv = rsqrtf(ss * (1.0f / DH2) + RMS_EPS) * (1.0f - LAMB_INIT);
#pragma unroll
  for (int i = 0; i < 8; ++i) {
    int d = lane * 8 + i;
    Ocat[(size_t)bn * (NHEADS * DH2) + hh * DH2 + d] = (_Float16)(vals[i] * inv * rms_scale[d]);
  }
}

// ---------------------------------------------------------------------------
extern "C" void kernel_launch(void* const* d_in, const int* in_sizes, int n_in,
                              void* d_out, int out_size, void* d_ws, size_t ws_size,
                              hipStream_t stream) {
  (void)in_sizes; (void)n_in; (void)out_size; (void)ws_size;
  const float* X   = (const float*)d_in[0];
  const float* Wq  = (const float*)d_in[1];
  const float* Wk  = (const float*)d_in[2];
  const float* Wv  = (const float*)d_in[3];
  const float* Wo  = (const float*)d_in[4];
  const float* lq1 = (const float*)d_in[5];
  const float* lk1 = (const float*)d_in[6];
  const float* lq2 = (const float*)d_in[7];
  const float* lk2 = (const float*)d_in[8];
  const float* rs  = (const float*)d_in[9];

  const size_t nX = (size_t)BATCH * SEQ_N * DMODEL;        // 8,388,608
  const size_t nW = (size_t)2 * DMODEL * DMODEL;           // 8,388,608
  const size_t nP = (size_t)BATCH * NHEADS * SEQ_N * DH2;  // 16,777,216

  _Float16* x16  = (_Float16*)d_ws;
  _Float16* wq16 = x16 + nX;
  _Float16* wk16 = wq16 + nW;
  _Float16* wv16 = wk16 + nW;
  _Float16* wo16 = wv16 + nW;
  _Float16* q16  = wo16 + nW;
  _Float16* k16  = q16 + nP;
  _Float16* vT16 = k16 + nP;
  _Float16* ocat = vT16 + nP;
  float* O1   = (float*)(ocat + nP);
  float* O2   = O1 + nP;
  float* lamp = O2 + nP;

  f32_to_f16<<<2048, 256, 0, stream>>>(X, x16, (int)(nX / 4));
  f32_to_f16<<<2048, 256, 0, stream>>>(Wq, wq16, (int)(nW / 4));
  f32_to_f16<<<2048, 256, 0, stream>>>(Wk, wk16, (int)(nW / 4));
  f32_to_f16<<<2048, 256, 0, stream>>>(Wv, wv16, (int)(nW / 4));
  f32_to_f16<<<2048, 256, 0, stream>>>(Wo, wo16, (int)(nW / 4));
  lam_kernel<<<1, 32, 0, stream>>>(lq1, lk1, lq2, lk2, lamp);

  // QKV projections: M=4096, Nout=4096, K=2048
  gemm_bt_wmma<<<dim3(32, 32), 256, 0, stream>>>(x16, wq16, q16, DMODEL, 0, 0);
  gemm_bt_wmma<<<dim3(32, 32), 256, 0, stream>>>(x16, wk16, k16, DMODEL, 0, 0);
  gemm_bt_wmma<<<dim3(32, 32), 256, 0, stream>>>(x16, wv16, vT16, DMODEL, 2, 0);

  // Two-component flash attention
  flash_component<<<dim3(BATCH * NHEADS, SEQ_N / 128, 2), 256, 0, stream>>>(
      q16, k16, vT16, O1, O2);

  // Differential combine + RMS norm
  combine_rms<<<dim3((BATCH * SEQ_N * NHEADS) / 8), 256, 0, stream>>>(O1, O2, lamp, rs, ocat);

  // Output projection: M=4096, Nout=2048, K=4096 -> f32 d_out
  gemm_bt_wmma<<<dim3(16, 32), 256, 0, stream>>>(ocat, wo16, d_out, 2 * DMODEL, 1, DMODEL);
}
